// AdvancedTrustTemporalGNN_76836964926237
// MI455X (gfx1250) — matrix operand order repaired and checked by reading, not compiled, not run on hardware
//
#include <hip/hip_runtime.h>

// ---------------------------------------------------------------------------
// AdvancedTrustTemporalGNN for MI455X (gfx1250, wave32, WMMA bf16,
// async global->LDS staging, L2-resident segment atomics)
// ---------------------------------------------------------------------------

constexpr int kN    = 16384;
constexpr int kE    = 131072;
constexpr int kInD  = 256;
constexpr int kEdD  = 128;
constexpr int kD    = 512;
constexpr int kH    = 8;
constexpr int kKVH  = 2;
constexpr int kHD   = 64;
constexpr int kKVD  = 128;
constexpr int kNEXP = 4;
constexpr int kL    = 2;

#ifndef USE_ASYNC_LDS
#define USE_ASYNC_LDS 1
#endif

typedef __attribute__((ext_vector_type(16))) __bf16 v16bf;
typedef __attribute__((ext_vector_type(8)))  float  v8f;

__device__ __forceinline__ __bf16 f2bf(float f) {
  unsigned u = __builtin_bit_cast(unsigned, f);
  u = (u + 0x7fffu + ((u >> 16) & 1u)) >> 16;
  unsigned short s = (unsigned short)u;
  return __builtin_bit_cast(__bf16, s);
}
__device__ __forceinline__ unsigned fenc(float f) {
  unsigned u = __builtin_bit_cast(unsigned, f);
  return (u & 0x80000000u) ? ~u : (u | 0x80000000u);
}
__device__ __forceinline__ float fdec(unsigned u) {
  unsigned v = (u & 0x80000000u) ? (u & 0x7fffffffu) : ~u;
  return __builtin_bit_cast(float, v);
}
__device__ __forceinline__ float sigmoidf(float x) { return 1.f / (1.f + __expf(-x)); }
__device__ __forceinline__ float geluf(float x) { return 0.5f * x * (1.f + erff(x * 0.70710678118654752f)); }
__device__ __forceinline__ float siluf(float x) { return x * sigmoidf(x); }

// CDNA5 async global->LDS copy (16 bytes per lane), ASYNCcnt-tracked.
__device__ __forceinline__ void cp_async_b128(void* lds, const void* g) {
#if USE_ASYNC_LDS
  unsigned loff = (unsigned)(unsigned long long)lds;   // LDS aperture low bits
  unsigned long long ga = (unsigned long long)g;
  asm volatile("global_load_async_to_lds_b128 %0, %1, off"
               :: "v"(loff), "v"(ga) : "memory");
#else
  *(uint4*)lds = *(const uint4*)g;
#endif
}
__device__ __forceinline__ void wait_async_all() {
#if USE_ASYNC_LDS
#if __has_builtin(__builtin_amdgcn_s_wait_asynccnt)
  __builtin_amdgcn_s_wait_asynccnt(0);
#else
  asm volatile("s_wait_asynccnt 0x0" ::: "memory");
#endif
#endif
}

// ---------------------------------------------------------------------------
// fp32 -> bf16 conversion pass (4 elements/thread)
// ---------------------------------------------------------------------------
__global__ void k_cvt_bf16(const float* __restrict__ x, __bf16* __restrict__ y)
{
  size_t i = ((size_t)blockIdx.x * 256 + threadIdx.x) * 4;
  float4 v = *(const float4*)(x + i);
  y[i + 0] = f2bf(v.x);
  y[i + 1] = f2bf(v.y);
  y[i + 2] = f2bf(v.z);
  y[i + 3] = f2bf(v.w);
}

// ---------------------------------------------------------------------------
// WMMA bf16 GEMM with double-buffered async LDS staging.
// C[M,N] = act(A[M,K] @ W[K,N] + bias)     (A, W pre-converted to bf16)
// mode 0: C = act(v);  mode 1: C += rowscale[m*rs_stride] * v
// Requires M%128==0, N%128==0, K%32==0 (true for every GEMM here).
// Block: 256 threads (8 waves), 128x128 tile; wave (mw,nw) owns 32x64.
// ---------------------------------------------------------------------------
__global__ void __launch_bounds__(256)
k_gemm_bf(const __bf16* __restrict__ A, const __bf16* __restrict__ W,
          const float* __restrict__ bias, float* __restrict__ C,
          int M, int K, int N, int act, int mode,
          const float* __restrict__ rowscale, int rs_stride)
{
  __shared__ __bf16 As[2][128 * 40];   // 128 rows x 32 cols, pad 8 (16B-aligned rows)
  __shared__ __bf16 Ws[2][32 * 136];   // 32 rows x 128 cols, pad 8

  const int tid  = threadIdx.x;
  const int lane = tid & 31;
  const int wv   = tid >> 5;          // 0..7
  const int mw   = wv & 3;            // M sub-tile: 32*mw
  const int nw   = wv >> 2;           // N sub-tile: 64*nw
  const int half = lane >> 4;
  const int l15  = lane & 15;
  const int m0   = blockIdx.y * 128;
  const int n0   = blockIdx.x * 128;

  v8f acc[2][4];
#pragma unroll
  for (int mi = 0; mi < 2; ++mi)
#pragma unroll
    for (int nt = 0; nt < 4; ++nt) {
      float b = bias ? bias[n0 + nw * 64 + nt * 16 + l15] : 0.f;
#pragma unroll
      for (int r = 0; r < 8; ++r) acc[mi][nt][r] = b;
    }

  const int T = K / 32;

  // stage tile `step` into buffer `buf`
  auto issue = [&](int buf, int k0) {
#pragma unroll
    for (int q = 0; q < 2; ++q) {
      int c = tid + q * 256;               // 0..511
      int ar = c >> 2, ac = (c & 3) * 8;   // A: 128 rows x 4 chunks
      cp_async_b128(&As[buf][ar * 40 + ac], A + (size_t)(m0 + ar) * K + k0 + ac);
      int wr = c >> 4, wc = (c & 15) * 8;  // W: 32 rows x 16 chunks
      cp_async_b128(&Ws[buf][wr * 136 + wc], W + (size_t)(k0 + wr) * N + n0 + wc);
    }
  };

  issue(0, 0);

  for (int i = 0; i < T; ++i) {
    wait_async_all();
    __syncthreads();                       // everyone's tile-i copies visible
    if (i + 1 < T) issue((i + 1) & 1, (i + 1) * 32);

    const int b = i & 1;
    v16bf af[2];
#pragma unroll
    for (int mi = 0; mi < 2; ++mi) {
      int row = mw * 32 + mi * 16 + l15;
#pragma unroll
      for (int r = 0; r < 8; ++r) {
        int kr = (r < 4 ? 2 * r : 2 * r + 8) + half * 8;
        af[mi][2 * r]     = As[b][row * 40 + kr];
        af[mi][2 * r + 1] = As[b][row * 40 + kr + 1];
      }
    }
#pragma unroll
    for (int nt = 0; nt < 4; ++nt) {
      v16bf bfr;
#pragma unroll
      for (int r = 0; r < 8; ++r) {
        int kr = (r < 4 ? 2 * r : 2 * r + 8) + half * 8;
        bfr[2 * r]     = Ws[b][kr * 136 + nw * 64 + nt * 16 + l15];
        bfr[2 * r + 1] = Ws[b][(kr + 1) * 136 + nw * 64 + nt * 16 + l15];
      }
      acc[0][nt] = __builtin_amdgcn_wmma_f32_16x16x32_bf16(
          false, af[0], false, bfr, (short)0, acc[0][nt], false, false);
      acc[1][nt] = __builtin_amdgcn_wmma_f32_16x16x32_bf16(
          false, af[1], false, bfr, (short)0, acc[1][nt], false, false);
    }
    __syncthreads();                       // done reading buf b before it is reused
  }

#pragma unroll
  for (int mi = 0; mi < 2; ++mi)
#pragma unroll
    for (int nt = 0; nt < 4; ++nt)
#pragma unroll
      for (int r = 0; r < 8; ++r) {
        int m = m0 + mw * 32 + mi * 16 + 8 * half + r;
        int n = n0 + nw * 64 + nt * 16 + l15;
        float v = acc[mi][nt][r];
        size_t idx = (size_t)m * N + n;
        if (mode == 1) {
          C[idx] += rowscale[(size_t)m * rs_stride] * v;
        } else {
          if (act == 1) v = geluf(v);
          C[idx] = v;
        }
      }
}

// ---------------------------------------------------------------------------
// mem_u = mem / (1 + max(sqrt(mem_var), 1e-6))
// ---------------------------------------------------------------------------
__global__ void k_prep_mem(const float* __restrict__ mem, const float* __restrict__ mvar,
                           float* __restrict__ memu)
{
  size_t i = (size_t)blockIdx.x * 256 + threadIdx.x;
  float u = 1.f / (1.f + fmaxf(sqrtf(mvar[i]), 1e-6f));
  memu[i] = mem[i] * u;
}

// ---------------------------------------------------------------------------
// In-place RoPE on X: [n, heads, 64]; one thread per (node, head, pair)
// ---------------------------------------------------------------------------
__global__ void k_rope(float* __restrict__ X, const float* __restrict__ ntime, int heads)
{
  size_t i = (size_t)blockIdx.x * 256 + threadIdx.x;
  int pair = (int)(i & 31);
  size_t rest = i >> 5;
  int head = (int)(rest % heads);
  size_t node = rest / heads;
  float t = ntime[node];
  float freq = t * __powf(10000.f, -(float)(2 * pair) / 64.f);
  float c = cosf(freq), s = sinf(freq);
  size_t base = node * (size_t)heads * 64 + (size_t)head * 64 + 2 * pair;
  float x1 = X[base], x2 = X[base + 1];
  X[base]     = x1 * c - x2 * s;
  X[base + 1] = x2 * c + x1 * s;
}

__global__ void k_fill_u32(unsigned* __restrict__ p, unsigned v)
{
  size_t i = (size_t)blockIdx.x * 256 + threadIdx.x;
  p[i] = v;
}

// ---------------------------------------------------------------------------
// Edge attention + WMMA-batched trust network.
// Block: 128 threads (4 waves); each wave handles 2 edges = 16 (edge,head) rows.
// ---------------------------------------------------------------------------
__global__ void __launch_bounds__(128)
k_edge_attn(const float* __restrict__ Qr,    // [N,8,64] rope applied
            const float* __restrict__ Kr,    // [N,2,64] rope applied
            const float* __restrict__ Ee,    // [E,2,64]
            const float* __restrict__ memu,  // [N,512]
            const int*   __restrict__ eidx,  // [2,E]
            const float* __restrict__ tq_w, const float* __restrict__ tq_b,
            const float* __restrict__ tk_w, const float* __restrict__ tk_b,
            const float* __restrict__ tout_w, const float* __restrict__ tout_b,
            const float* __restrict__ log_temp,
            float* __restrict__ attn,        // [E,8]
            unsigned* __restrict__ mmax)     // [N,8] encoded floats
{
  __shared__ __bf16 WQ[128 * 68];
  __shared__ __bf16 WK[128 * 68];
  __shared__ __bf16 rowsQ[4][16 * 136];
  __shared__ __bf16 rowsK[4][16 * 136];
  __shared__ float  araw[4][16];

  const int tid  = threadIdx.x;
  const int lane = tid & 31;
  const int wv   = tid >> 5;
  const int half = lane >> 4;
  const int l15  = lane & 15;
  const int e0   = blockIdx.x * 8 + wv * 2;

  for (int i = tid; i < 128 * 64; i += 128) {
    int k = i >> 6, n = i & 63;
    WQ[k * 68 + n] = f2bf(tq_w[i]);
    WK[k * 68 + n] = f2bf(tk_w[i]);
  }

  {
    int r  = lane >> 1;
    int hf = lane & 1;
    int e  = e0 + (r >> 3);
    int hh = r & 7;
    int kvh = hh >> 2;
    int src = eidx[e];
    int dst = eidx[kE + e];
    float dot = 0.f;
#pragma unroll 4
    for (int c = 0; c < 64; ++c) {
      int col = hf * 64 + c;
      float qv, kvv;
      if (hf == 0) {
        qv  = memu[(size_t)dst * kD + hh * 64 + c];
        kvv = memu[(size_t)src * kD + hh * 64 + c];
      } else {
        qv  = Qr[(size_t)dst * kD + hh * 64 + c];
        kvv = Kr[(size_t)src * kKVD + kvh * 64 + c] + Ee[(size_t)e * kKVD + kvh * 64 + c];
        dot += qv * kvv;
      }
      rowsQ[wv][r * 136 + col] = f2bf(qv);
      rowsK[wv][r * 136 + col] = f2bf(kvv);
    }
    if (hf == 1) araw[wv][r] = dot * 0.125f / __expf(log_temp[hh]);
  }
  __syncthreads();

  v16bf aq[4], ak[4];
#pragma unroll
  for (int ks = 0; ks < 4; ++ks) {
#pragma unroll
    for (int r = 0; r < 8; ++r) {
      int kr = ks * 32 + (r < 4 ? 2 * r : 2 * r + 8) + half * 8;
      aq[ks][2 * r]     = rowsQ[wv][l15 * 136 + kr];
      aq[ks][2 * r + 1] = rowsQ[wv][l15 * 136 + kr + 1];
      ak[ks][2 * r]     = rowsK[wv][l15 * 136 + kr];
      ak[ks][2 * r + 1] = rowsK[wv][l15 * 136 + kr + 1];
    }
  }

  v8f qac[4], kac[4];
#pragma unroll
  for (int nt = 0; nt < 4; ++nt) {
    float bq = tq_b[nt * 16 + l15];
    float bk = tk_b[nt * 16 + l15];
#pragma unroll
    for (int r = 0; r < 8; ++r) { qac[nt][r] = bq; kac[nt][r] = bk; }
  }

#pragma unroll
  for (int nt = 0; nt < 4; ++nt) {
#pragma unroll
    for (int ks = 0; ks < 4; ++ks) {
      v16bf bq, bk;
#pragma unroll
      for (int r = 0; r < 8; ++r) {
        int kk = ks * 32 + (r < 4 ? 2 * r : 2 * r + 8) + half * 8;
        bq[2 * r]     = WQ[kk * 68 + nt * 16 + l15];
        bq[2 * r + 1] = WQ[(kk + 1) * 68 + nt * 16 + l15];
        bk[2 * r]     = WK[kk * 68 + nt * 16 + l15];
        bk[2 * r + 1] = WK[(kk + 1) * 68 + nt * 16 + l15];
      }
      qac[nt] = __builtin_amdgcn_wmma_f32_16x16x32_bf16(false, aq[ks], false, bq, (short)0, qac[nt], false, false);
      kac[nt] = __builtin_amdgcn_wmma_f32_16x16x32_bf16(false, ak[ks], false, bk, (short)0, kac[nt], false, false);
    }
  }

  float p[8];
#pragma unroll
  for (int r = 0; r < 8; ++r) p[r] = 0.f;
#pragma unroll
  for (int nt = 0; nt < 4; ++nt) {
    float tw = tout_w[nt * 16 + l15];
#pragma unroll
    for (int r = 0; r < 8; ++r)
      p[r] += tanhf(qac[nt][r] * kac[nt][r]) * tw;
  }
#pragma unroll
  for (int r = 0; r < 8; ++r) {
    for (int m = 1; m < 16; m <<= 1) p[r] += __shfl_xor(p[r], m, 32);
  }

  if (l15 == 0) {
    float tb = tout_b[0];
#pragma unroll
    for (int r = 0; r < 8; ++r) {
      int m  = 8 * half + r;
      int e  = e0 + (m >> 3);
      int hh = m & 7;
      float trust = sigmoidf(p[r] + tb);
      float a = araw[wv][m] * trust;
      attn[(size_t)e * kH + hh] = a;
      int dst = eidx[kE + e];
      atomicMax(&mmax[(size_t)dst * kH + hh], fenc(a));
    }
  }
}

// ---------------------------------------------------------------------------
__global__ void k_expsum(float* __restrict__ attn, const unsigned* __restrict__ mmax,
                         const int* __restrict__ eidx, float* __restrict__ ssum)
{
  size_t i = (size_t)blockIdx.x * 256 + threadIdx.x;   // E*H
  int e = (int)(i >> 3), hh = (int)(i & 7);
  int dst = eidx[kE + e];
  float ex = __expf(attn[i] - fdec(mmax[(size_t)dst * kH + hh]));
  attn[i] = ex;
  atomicAdd(&ssum[(size_t)dst * kH + hh], ex);
}

// ---------------------------------------------------------------------------
__global__ void k_agg(const float* __restrict__ attn, const float* __restrict__ ssum,
                      const float* __restrict__ V, const int* __restrict__ eidx,
                      float* __restrict__ out)
{
  size_t i = (size_t)blockIdx.x * 256 + threadIdx.x;   // E*512
  int e = (int)(i >> 9);
  int rem = (int)(i & 511);
  int hh = rem >> 6, c = rem & 63;
  int src = eidx[e], dst = eidx[kE + e];
  float a = attn[(size_t)e * kH + hh] / ssum[(size_t)dst * kH + hh];
  atomicAdd(&out[(size_t)dst * kD + rem], V[(size_t)src * kKVD + (hh >> 2) * 64 + c] * a);
}

// ---------------------------------------------------------------------------
__global__ void k_gate(const float* __restrict__ out, const float* __restrict__ gate_w,
                       const float* __restrict__ gate_b, float* __restrict__ wk)
{
  int n = blockIdx.x * 256 + threadIdx.x;
  float lg[4] = {gate_b[0], gate_b[1], gate_b[2], gate_b[3]};
  for (int k = 0; k < kD; ++k) {
    float hv = out[(size_t)n * kD + k];
#pragma unroll
    for (int j = 0; j < 4; ++j) lg[j] += hv * gate_w[k * 4 + j];
  }
  int i0 = 0;
#pragma unroll
  for (int j = 1; j < 4; ++j) if (lg[j] > lg[i0]) i0 = j;
  int i1 = -1;
#pragma unroll
  for (int j = 0; j < 4; ++j) if (j != i0 && (i1 < 0 || lg[j] > lg[i1])) i1 = j;
  float e1 = __expf(lg[i1] - lg[i0]);
  float w0 = 1.f / (1.f + e1);
  float w1 = e1 / (1.f + e1);
#pragma unroll
  for (int j = 0; j < 4; ++j)
    wk[(size_t)n * 4 + j] = (j == i0) ? w0 : ((j == i1) ? w1 : 0.f);
}

// ---------------------------------------------------------------------------
__global__ void k_colsum(const float* __restrict__ ada_w, float* __restrict__ colsum)
{
  int j = blockIdx.x * 256 + threadIdx.x;   // 1024
  float s = 0.f;
  for (int k = 0; k < kD; ++k) s += ada_w[(size_t)k * 1024 + j];
  colsum[j] = s;
}

__device__ __forceinline__ float blockSum(float v, float* red)
{
  int t = threadIdx.x;
  red[t] = v;
  __syncthreads();
  for (int s = 128; s > 0; s >>= 1) {
    if (t < s) red[t] += red[t + s];
    __syncthreads();
  }
  float r = red[0];
  __syncthreads();
  return r;
}

// ---------------------------------------------------------------------------
__global__ void __launch_bounds__(256)
k_node_final(const float* __restrict__ h, const float* __restrict__ o,
             const float* __restrict__ ntime, const float* __restrict__ colsum,
             const float* __restrict__ ada_b, const float* __restrict__ rg_w,
             const float* __restrict__ rg_b, float* __restrict__ hout)
{
  __shared__ float red[256];
  const int n = blockIdx.x;
  const int t = threadIdx.x;
  const size_t base = (size_t)n * kD;

  float bdot = h[base + t] * rg_w[t] + h[base + t + 256] * rg_w[t + 256];
  float beta = sigmoidf(blockSum(bdot, red) + rg_b[0]);

  float o0 = o[base + t], o1 = o[base + t + 256];
  float mu  = blockSum(o0 + o1, red) * (1.f / kD);
  float msq = blockSum(o0 * o0 + o1 * o1, red) * (1.f / kD);
  float rstd = rsqrtf(msq - mu * mu + 1e-5f);
  float st = siluf(ntime[n]);

#pragma unroll
  for (int q = 0; q < 2; ++q) {
    int c = t + q * 256;
    float scale = st * colsum[c] + ada_b[c];
    float shift = st * colsum[512 + c] + ada_b[512 + c];
    float ov = (q ? o1 : o0);
    float on = (ov - mu) * rstd * (1.f + scale) + shift;
    float hv = h[base + c];
    hout[base + c] = beta * hv + (1.f - beta) * on;
  }
}

// ---------------------------------------------------------------------------
__global__ void __launch_bounds__(256)
k_final_ln(const float* __restrict__ h, const float* __restrict__ g,
           const float* __restrict__ b, float* __restrict__ out)
{
  __shared__ float red[256];
  const int n = blockIdx.x;
  const int t = threadIdx.x;
  const size_t base = (size_t)n * kD;
  float h0 = h[base + t], h1 = h[base + t + 256];
  float mu  = blockSum(h0 + h1, red) * (1.f / kD);
  float msq = blockSum(h0 * h0 + h1 * h1, red) * (1.f / kD);
  float rstd = rsqrtf(msq - mu * mu + 1e-5f);
  out[base + t]       = (h0 - mu) * rstd * g[t] + b[t];
  out[base + t + 256] = (h1 - mu) * rstd * g[t + 256] + b[t + 256];
}

// ---------------------------------------------------------------------------
// Host launcher
// ---------------------------------------------------------------------------
extern "C" void kernel_launch(void* const* d_in, const int* in_sizes, int n_in,
                              void* d_out, int out_size, void* d_ws, size_t ws_size,
                              hipStream_t stream)
{
  (void)in_sizes; (void)n_in; (void)out_size; (void)ws_size;

  const float* x        = (const float*)d_in[0];
  const float* ntime    = (const float*)d_in[1];
  const float* edge_attr= (const float*)d_in[2];
  const int*   eidx     = (const int*)  d_in[3];
  const float* mem      = (const float*)d_in[4];
  const float* mvar     = (const float*)d_in[5];
  const float* in_w     = (const float*)d_in[6];
  const float* in_b     = (const float*)d_in[7];
  const float* edge_w   = (const float*)d_in[8];
  const float* edge_b   = (const float*)d_in[9];
  const float* q_w      = (const float*)d_in[10];
  const float* q_b      = (const float*)d_in[11];
  const float* k_w      = (const float*)d_in[12];
  const float* k_b      = (const float*)d_in[13];
  const float* v_w      = (const float*)d_in[14];
  const float* v_b      = (const float*)d_in[15];
  const float* o_w      = (const float*)d_in[16];
  const float* o_b      = (const float*)d_in[17];
  const float* e_w      = (const float*)d_in[18];
  const float* e_b      = (const float*)d_in[19];
  const float* log_temp = (const float*)d_in[20];
  const float* tq_w     = (const float*)d_in[21];
  const float* tq_b     = (const float*)d_in[22];
  const float* tk_w     = (const float*)d_in[23];
  const float* tk_b     = (const float*)d_in[24];
  const float* tout_w   = (const float*)d_in[25];
  const float* tout_b   = (const float*)d_in[26];
  const float* gate_w   = (const float*)d_in[27];
  const float* gate_b   = (const float*)d_in[28];
  const float* exp_w1   = (const float*)d_in[29];
  const float* exp_b1   = (const float*)d_in[30];
  const float* exp_w2   = (const float*)d_in[31];
  const float* exp_b2   = (const float*)d_in[32];
  const float* ada_w    = (const float*)d_in[33];
  const float* ada_b    = (const float*)d_in[34];
  const float* rg_w     = (const float*)d_in[35];
  const float* rg_b     = (const float*)d_in[36];
  const float* fn_g     = (const float*)d_in[37];
  const float* fn_b     = (const float*)d_in[38];

  char* wsp = (char*)d_ws;
  auto allocf = [&](size_t elems) -> float* {
    float* p = (float*)wsp;
    wsp += ((elems * sizeof(float) + 255) & ~size_t(255));
    return p;
  };
  auto allocb = [&](size_t elems) -> __bf16* {
    __bf16* p = (__bf16*)wsp;
    wsp += ((elems * sizeof(short) + 255) & ~size_t(255));
    return p;
  };

  // fp32 buffers
  float* hA     = allocf((size_t)kN * kD);
  float* hB     = allocf((size_t)kN * kD);
  float* edge_h = allocf((size_t)kE * kD);
  float* Qr     = allocf((size_t)kN * kD);
  float* Kr     = allocf((size_t)kN * kKVD);
  float* Vv     = allocf((size_t)kN * kKVD);
  float* Ee     = allocf((size_t)kE * kKVD);
  float* memu   = allocf((size_t)kN * kD);
  float* attn   = allocf((size_t)kE * kH);
  unsigned* mmax = (unsigned*)allocf((size_t)kN * kH);
  float* ssum   = allocf((size_t)kN * kH);
  float* agg    = allocf((size_t)kN * kD);
  float* hdn    = allocf((size_t)kN * 2 * kD);
  float* moe    = allocf((size_t)kN * kD);
  float* obuf   = allocf((size_t)kN * kD);
  float* wk     = allocf((size_t)kN * kNEXP);
  float* colsum = allocf(2 * kD);

  // bf16 operand buffers
  __bf16* xb    = allocb((size_t)kN * kInD);
  __bf16* eab   = allocb((size_t)kE * kEdD);
  __bf16* hb    = allocb((size_t)kN * kD);
  __bf16* ehb   = allocb((size_t)kE * kD);
  __bf16* aggb  = allocb((size_t)kN * kD);
  __bf16* hdnb  = allocb((size_t)kN * 2 * kD);
  __bf16* moeb  = allocb((size_t)kN * kD);
  __bf16* inwb  = allocb((size_t)kInD * kD);
  __bf16* edwb  = allocb((size_t)kEdD * kD);
  __bf16* qwb   = allocb((size_t)kL * kD * kD);
  __bf16* kwb   = allocb((size_t)kL * kD * kKVD);
  __bf16* vwb   = allocb((size_t)kL * kD * kKVD);
  __bf16* ewb   = allocb((size_t)kL * kD * kKVD);
  __bf16* owb   = allocb((size_t)kL * kD * kD);
  __bf16* w1b   = allocb((size_t)kL * kNEXP * kD * 2 * kD);
  __bf16* w2b   = allocb((size_t)kL * kNEXP * 2 * kD * kD);

  auto cvt = [&](const float* src, __bf16* dst, size_t n) {
    k_cvt_bf16<<<n / 1024, 256, 0, stream>>>(src, dst);
  };
  auto gemm = [&](const __bf16* A, const __bf16* W, const float* B, float* C,
                  int M, int K, int N, int act, int mode,
                  const float* rs, int rstride) {
    dim3 g(N / 128, M / 128);
    k_gemm_bf<<<g, dim3(256), 0, stream>>>(A, W, B, C, M, K, N, act, mode, rs, rstride);
  };

  // one-time conversions
  cvt(x, xb, (size_t)kN * kInD);
  cvt(edge_attr, eab, (size_t)kE * kEdD);
  cvt(in_w, inwb, (size_t)kInD * kD);
  cvt(edge_w, edwb, (size_t)kEdD * kD);
  cvt(q_w, qwb, (size_t)kL * kD * kD);
  cvt(k_w, kwb, (size_t)kL * kD * kKVD);
  cvt(v_w, vwb, (size_t)kL * kD * kKVD);
  cvt(e_w, ewb, (size_t)kL * kD * kKVD);
  cvt(o_w, owb, (size_t)kL * kD * kD);
  cvt(exp_w1, w1b, (size_t)kL * kNEXP * kD * 2 * kD);
  cvt(exp_w2, w2b, (size_t)kL * kNEXP * 2 * kD * kD);

  // prep
  k_prep_mem<<<(size_t)kN * kD / 256, 256, 0, stream>>>(mem, mvar, memu);
  gemm(xb, inwb, in_b, hA, kN, kInD, kD, 0, 0, nullptr, 0);
  gemm(eab, edwb, edge_b, edge_h, kE, kEdD, kD, 0, 0, nullptr, 0);
  cvt(edge_h, ehb, (size_t)kE * kD);

  float* hcur = hA;
  float* hnxt = hB;

  for (int li = 0; li < kL; ++li) {
    const __bf16* qw = qwb + (size_t)li * kD * kD;
    const __bf16* kw = kwb + (size_t)li * kD * kKVD;
    const __bf16* vw = vwb + (size_t)li * kD * kKVD;
    const __bf16* ow = owb + (size_t)li * kD * kD;
    const __bf16* ew = ewb + (size_t)li * kD * kKVD;
    const float* qb  = q_b  + (size_t)li * kD;
    const float* kb  = k_b  + (size_t)li * kKVD;
    const float* vb  = v_b  + (size_t)li * kKVD;
    const float* ob  = o_b  + (size_t)li * kD;
    const float* eb  = e_b  + (size_t)li * kKVD;
    const float* lt  = log_temp + (size_t)li * kH;
    const float* tqw = tq_w + (size_t)li * 2 * kHD * kHD;
    const float* tqb = tq_b + (size_t)li * kHD;
    const float* tkw = tk_w + (size_t)li * 2 * kHD * kHD;
    const float* tkb = tk_b + (size_t)li * kHD;
    const float* tow = tout_w + (size_t)li * kHD;
    const float* tob = tout_b + (size_t)li;
    const float* gw  = gate_w + (size_t)li * kD * kNEXP;
    const float* gb  = gate_b + (size_t)li * kNEXP;
    const float* adw = ada_w + (size_t)li * kD * 2 * kD;
    const float* adb = ada_b + (size_t)li * 2 * kD;
    const float* rgw = rg_w + (size_t)li * kD;
    const float* rgb = rg_b + (size_t)li;

    // projections + RoPE
    cvt(hcur, hb, (size_t)kN * kD);
    gemm(hb, qw, qb, Qr, kN, kD, kD, 0, 0, nullptr, 0);
    gemm(hb, kw, kb, Kr, kN, kD, kKVD, 0, 0, nullptr, 0);
    gemm(hb, vw, vb, Vv, kN, kD, kKVD, 0, 0, nullptr, 0);
    gemm(ehb, ew, eb, Ee, kE, kD, kKVD, 0, 0, nullptr, 0);
    k_rope<<<(size_t)kN * kH * 32 / 256, 256, 0, stream>>>(Qr, ntime, kH);
    k_rope<<<(size_t)kN * kKVH * 32 / 256, 256, 0, stream>>>(Kr, ntime, kKVH);

    // attention over edges
    k_fill_u32<<<(size_t)kN * kH / 256, 256, 0, stream>>>(mmax, 0x007fffffu); // enc(-inf)
    hipMemsetAsync(ssum, 0, (size_t)kN * kH * sizeof(float), stream);
    hipMemsetAsync(agg, 0, (size_t)kN * kD * sizeof(float), stream);
    k_edge_attn<<<kE / 8, 128, 0, stream>>>(Qr, Kr, Ee, memu, eidx,
                                            tqw, tqb, tkw, tkb, tow, tob, lt,
                                            attn, mmax);
    k_expsum<<<(size_t)kE * kH / 256, 256, 0, stream>>>(attn, mmax, eidx, ssum);
    k_agg<<<(size_t)kE * kD / 256, 256, 0, stream>>>(attn, ssum, Vv, eidx, agg);

    // MoE
    k_gate<<<kN / 256, 256, 0, stream>>>(agg, gw, gb, wk);
    hipMemsetAsync(moe, 0, (size_t)kN * kD * sizeof(float), stream);
    cvt(agg, aggb, (size_t)kN * kD);
    for (int ei = 0; ei < kNEXP; ++ei) {
      const __bf16* w1 = w1b + ((size_t)li * kNEXP + ei) * kD * 2 * kD;
      const __bf16* w2 = w2b + ((size_t)li * kNEXP + ei) * 2 * kD * kD;
      const float* b1 = exp_b1 + ((size_t)li * kNEXP + ei) * 2 * kD;
      const float* b2 = exp_b2 + ((size_t)li * kNEXP + ei) * kD;
      gemm(aggb, w1, b1, hdn, kN, kD, 2 * kD, 1, 0, nullptr, 0);      // GELU
      cvt(hdn, hdnb, (size_t)kN * 2 * kD);
      gemm(hdnb, w2, b2, moe, kN, 2 * kD, kD, 0, 1, wk + ei, kNEXP);  // += wk*eo
    }
    cvt(moe, moeb, (size_t)kN * kD);
    gemm(moeb, ow, ob, obuf, kN, kD, kD, 0, 0, nullptr, 0);

    // adaLN + residual gate
    k_colsum<<<2 * kD / 256, 256, 0, stream>>>(adw, colsum);
    k_node_final<<<kN, 256, 0, stream>>>(hcur, obuf, ntime, colsum, adb, rgw, rgb, hnxt);

    float* tmp = hcur; hcur = hnxt; hnxt = tmp;
  }

  k_final_ln<<<kN, 256, 0, stream>>>(hcur, fn_g, fn_b, (float*)d_out);
}
